// MultiHeadAttention_37606733643791
// MI455X (gfx1250) — matrix operand order repaired and checked by reading, not compiled, not run on hardware
//
#include <hip/hip_runtime.h>
#include <hip/hip_bf16.h>

#define BB 64
#define TT 256
#define CC 384
#define HH 6
#define HD 64

typedef __attribute__((ext_vector_type(16))) _Float16 v16h;
typedef __attribute__((ext_vector_type(8)))  float    v8f;

union FragU { unsigned int u[8]; v16h h; };

__device__ __forceinline__ v8f vzero8() {
  v8f z = {0.f, 0.f, 0.f, 0.f, 0.f, 0.f, 0.f, 0.f};
  return z;
}

__device__ __forceinline__ v8f wmma16(v16h a, v16h b, v8f c) {
  // v_wmma_f32_16x16x32_f16: D = A(16x32) * B(32x16) + C
  return __builtin_amdgcn_wmma_f32_16x16x32_f16(false, a, false, b, (short)0, c,
                                                false, false);
}

// A fragment: 16x32 f16 tile from row-major matrix (row stride = ld elements).
// ISA layout: lane m (0-15 per half), K = (vgpr<4?0:16) + 8*half + 2*(vgpr&3)+e
__device__ __forceinline__ v16h frag_a(const _Float16* base, int ld) {
  const int lane = threadIdx.x & 31;
  const int m = lane & 15, half = lane >> 4;
  const _Float16* row = base + (size_t)m * ld + 8 * half;
  FragU f;
#pragma unroll
  for (int j = 0; j < 8; ++j) {
    const int kk = ((j & 4) ? 16 : 0) + 2 * (j & 3);
    f.u[j] = *(const unsigned int*)(row + kk);
  }
  return f.h;
}

// B fragment: 32x16 f16 tile; source is the TRANSPOSED matrix Bt (Bt[n][k] = B[k][n]),
// row-major with row stride ld. ISA layout: lane n, K = (vgpr<4?0:16) + 8*half + 2*(vgpr&3)+e.
__device__ __forceinline__ v16h frag_b(const _Float16* base, int ld) {
  const int lane = threadIdx.x & 31;
  const int n = lane & 15, half = lane >> 4;
  const _Float16* row = base + (size_t)n * ld + 8 * half;
  FragU f;
#pragma unroll
  for (int j = 0; j < 8; ++j) f.u[j] = *(const unsigned int*)(row + ((j & 4) ? 16 : 0) + 2 * (j & 3));
  return f.h;
}

// Async global->LDS stage of a 2D slab (rows of row_elems f16, copied in 16B
// chunks) using GLOBAL_LOAD_ASYNC_TO_LDS_B128 (ASYNCcnt path). 256 threads,
// ITERS chunks per thread, CPR 16B-chunks per row. Ends with wait + barrier.
template <int ITERS, int CPR>
__device__ __forceinline__ void stage_async(const _Float16* __restrict__ g,
                                            _Float16* l, int ld_src, int ld_dst) {
  const int tid = threadIdx.x;
#pragma unroll
  for (int i = 0; i < ITERS; ++i) {
    const int c = tid + 256 * i;
    const int row = c / CPR;
    const int part = (c % CPR) * 8;
    const unsigned lo = (unsigned)(size_t)(l + row * ld_dst + part);
    const void* gp = (const void*)(g + (size_t)row * ld_src + part);
    asm volatile("global_load_async_to_lds_b128 %0, %1, off"
                 :: "v"(lo), "v"(gp) : "memory");
  }
  asm volatile("s_wait_asynccnt 0" ::: "memory");
  __syncthreads();
}

// ---------------- conversion / packing ----------------

__global__ void k_cvt_f32_f16(const float* __restrict__ src,
                              _Float16* __restrict__ dst, int n) {
  int i = blockIdx.x * blockDim.x + threadIdx.x;
  if (i < n) dst[i] = (_Float16)src[i];
}

// src [H][C][HD] fp32 -> dst [H][HD][C] f16
__global__ void k_transpose_w(const float* __restrict__ src,
                              _Float16* __restrict__ dst) {
  int i = blockIdx.x * blockDim.x + threadIdx.x;
  if (i >= HH * CC * HD) return;
  int h = i / (CC * HD);
  int r = i - h * CC * HD;
  int c = r / HD;
  int d = r - c * HD;
  dst[((size_t)h * HD + d) * CC + c] = (_Float16)src[i];
}

// ---------------- QKV projection ----------------
// grid = B * (T/16); block = 256 (8 waves). The 16-row x tile is async-staged
// to LDS once; each wave computes 9 of the 72 16x16 output tiles.
__global__ __launch_bounds__(256) void k_qkv(const _Float16* __restrict__ xh,
                                             const _Float16* __restrict__ wqt,
                                             const _Float16* __restrict__ wkt,
                                             const _Float16* __restrict__ wvt,
                                             _Float16* __restrict__ qh,
                                             _Float16* __restrict__ kh,
                                             _Float16* __restrict__ vt) {
  __shared__ __align__(16) _Float16 xls[16 * (CC + 8)];  // 12.25 KB, padded rows
  const int blk = blockIdx.x;
  const int b = blk >> 4;
  const int mt = blk & 15;
  const int R = mt * 16;
  const int w = threadIdx.x >> 5;
  const int lane = threadIdx.x & 31;
  const int n = lane & 15, half = lane >> 4;

  stage_async<3, 48>(xh + ((size_t)b * TT + R) * CC, xls, CC, CC + 8);

  int th[9], twhich[9], tnt[9];
  const _Float16* bptr[9];
#pragma unroll
  for (int i = 0; i < 9; ++i) {
    int tid = w + 8 * i;           // 0..71
    int h = tid / 12;
    int rem = tid - h * 12;
    int which = rem >> 2;          // 0=q 1=k 2=v
    int nt = rem & 3;
    th[i] = h; twhich[i] = which; tnt[i] = nt;
    const _Float16* wsrc = (which == 0) ? wqt : ((which == 1) ? wkt : wvt);
    bptr[i] = wsrc + ((size_t)h * HD + nt * 16) * CC;
  }
  v8f acc[9];
#pragma unroll
  for (int i = 0; i < 9; ++i) acc[i] = vzero8();

  for (int kt = 0; kt < 12; ++kt) {
    v16h a = frag_a(xls + kt * 32, CC + 8);
#pragma unroll
    for (int i = 0; i < 9; ++i) {
      v16h bf = frag_b(bptr[i] + kt * 32, CC);
      acc[i] = wmma16(a, bf, acc[i]);
    }
  }

#pragma unroll
  for (int i = 0; i < 9; ++i) {
    const int h = th[i], which = twhich[i], nt = tnt[i];
#pragma unroll
    for (int r = 0; r < 8; ++r) {
      const int row = r + 8 * half;
      const _Float16 val = (_Float16)acc[i][r];
      if (which == 0)
        qh[(((size_t)b * HH + h) * TT + R + row) * HD + nt * 16 + n] = val;
      else if (which == 1)
        kh[(((size_t)b * HH + h) * TT + R + row) * HD + nt * 16 + n] = val;
      else
        vt[(((size_t)b * HH + h) * HD + nt * 16 + n) * TT + R + row] = val;
    }
  }
}

// ---------------- attention core ----------------
// grid = B*H*2; block = 256 (8 waves). K slab (256x64 f16) async-staged to
// LDS, scores with causal tile-skipping (wave-uniform branch), softmax via
// cross-lane shuffles, then V slab re-staged into the same LDS buffer and
// P @ V with LDS-transposed P fragments.
__global__ __launch_bounds__(256) void k_attn(const _Float16* __restrict__ qh,
                                              const _Float16* __restrict__ kh,
                                              const _Float16* __restrict__ vt,
                                              _Float16* __restrict__ attnh) {
  __shared__ __align__(16) _Float16 kvls[256 * (HD + 8)];  // 36 KB (also holds V: 64*(TT+8))
  __shared__ __align__(16) _Float16 plds[8][16 * 40];      // 10 KB, padded stride
  const int blk = blockIdx.x;
  const int split = blk & 1;
  const int bh = blk >> 1;   // b*H + h
  const int h = bh % HH;
  const int b = bh / HH;
  const int w = threadIdx.x >> 5;
  const int lane = threadIdx.x & 31;
  const int n = lane & 15, half = lane >> 4;
  const int mt = 2 * w + split;      // interleaved for intra-block balance
  const int t0 = mt * 16;

  // stage K [256][64] -> LDS rows padded to 72
  stage_async<8, 8>(kh + (size_t)bh * TT * HD, kvls, HD, HD + 8);

  const _Float16* qbase = qh + ((size_t)bh * TT + t0) * HD;
  const v16h qa0 = frag_a(qbase, HD);
  const v16h qa1 = frag_a(qbase + 32, HD);

  v8f S[16];
#pragma unroll
  for (int j = 0; j < 16; ++j) {
    if (j <= mt) {                   // causal: skip fully-masked tiles
      const _Float16* kbase = kvls + j * 16 * (HD + 8);
      v16h kb0 = frag_b(kbase, HD + 8);
      v16h kb1 = frag_b(kbase + 32, HD + 8);
      v8f z = vzero8();
      z = wmma16(qa0, kb0, z);
      z = wmma16(qa1, kb1, z);
      S[j] = z;
    } else {
      S[j] = vzero8();
    }
  }

  const float scale = 0.125f;  // HD^-0.5
  float rinv[8];
#pragma unroll
  for (int r = 0; r < 8; ++r) {
    const int t = t0 + r + 8 * half;
    float pm = -1e30f;
#pragma unroll
    for (int j = 0; j < 16; ++j) {
      const int s = j * 16 + n;
      float val = S[j][r] * scale;
      if (s > t) val = -1e30f;          // causal mask
      S[j][r] = val;
      pm = fmaxf(pm, val);
    }
    // row max over the 16 lanes of this half (xor<16 stays in the group)
    pm = fmaxf(pm, __shfl_xor(pm, 1));
    pm = fmaxf(pm, __shfl_xor(pm, 2));
    pm = fmaxf(pm, __shfl_xor(pm, 4));
    pm = fmaxf(pm, __shfl_xor(pm, 8));
    float sum = 0.f;
#pragma unroll
    for (int j = 0; j < 16; ++j) {
      float p = __expf(S[j][r] - pm);
      S[j][r] = p;
      sum += p;
    }
    sum += __shfl_xor(sum, 1);
    sum += __shfl_xor(sum, 2);
    sum += __shfl_xor(sum, 4);
    sum += __shfl_xor(sum, 8);
    rinv[r] = 1.0f / sum;
  }

  // all waves done reading K from LDS; stage V [64][256] into the same buffer
  __syncthreads();
  stage_async<8, 32>(vt + (size_t)bh * HD * TT, kvls, TT, TT + 8);

  _Float16* myp = &plds[w][0];
  v8f O[4];
#pragma unroll
  for (int nt = 0; nt < 4; ++nt) O[nt] = vzero8();

#pragma unroll
  for (int sb = 0; sb < 8; ++sb) {
    if (2 * sb <= mt) {              // causal: chunk contributes only below diag
      // C-layout -> LDS (16 rows x 32 cols of P, f16, padded stride 40)
#pragma unroll
      for (int jj = 0; jj < 2; ++jj) {
        const int j = 2 * sb + jj;
#pragma unroll
        for (int r = 0; r < 8; ++r) {
          myp[(r + 8 * half) * 40 + jj * 16 + n] = (_Float16)S[j][r];
        }
      }
      asm volatile("s_wait_dscnt 0" ::: "memory");  // per-wave LDS RAW fence
      v16h pa = frag_a(myp, 40);
#pragma unroll
      for (int nt = 0; nt < 4; ++nt) {
        const _Float16* vbase = kvls + (nt * 16) * (TT + 8) + sb * 32;
        v16h vb = frag_b(vbase, TT + 8);
        O[nt] = wmma16(pa, vb, O[nt]);
      }
    }
  }

#pragma unroll
  for (int nt = 0; nt < 4; ++nt) {
#pragma unroll
    for (int r = 0; r < 8; ++r) {
      const int t = t0 + r + 8 * half;
      attnh[((size_t)b * TT + t) * CC + h * HD + nt * 16 + n] =
          (_Float16)(O[nt][r] * rinv[r]);
    }
  }
}

// ---------------- output projection ----------------
// grid = (B*T)/16; block = 256. A-tile async-staged to LDS; wave w handles
// n-tiles {w, w+8, w+16}.
__global__ __launch_bounds__(256) void k_proj(const _Float16* __restrict__ attnh,
                                              const _Float16* __restrict__ wph,
                                              const float* __restrict__ bp,
                                              float* __restrict__ out) {
  __shared__ __align__(16) _Float16 als[16 * (CC + 8)];
  const size_t R = (size_t)blockIdx.x * 16;  // global row over B*T
  const int w = threadIdx.x >> 5;
  const int lane = threadIdx.x & 31;
  const int n = lane & 15, half = lane >> 4;

  stage_async<3, 48>(attnh + R * CC, als, CC, CC + 8);

  v8f acc[3];
#pragma unroll
  for (int i = 0; i < 3; ++i) acc[i] = vzero8();

  for (int kt = 0; kt < 12; ++kt) {
    v16h a = frag_a(als + kt * 32, CC + 8);
#pragma unroll
    for (int i = 0; i < 3; ++i) {
      const int nt = w + 8 * i;
      v16h bf = frag_b(wph + (size_t)(nt * 16) * CC + kt * 32, CC);
      acc[i] = wmma16(a, bf, acc[i]);
    }
  }

#pragma unroll
  for (int i = 0; i < 3; ++i) {
    const int nt = w + 8 * i;
    const int col = nt * 16 + n;
    const float bias = bp[col];
#pragma unroll
    for (int r = 0; r < 8; ++r) {
      out[(R + r + 8 * half) * CC + col] = acc[i][r] + bias;
    }
  }
}

// ---------------- host launcher ----------------
extern "C" void kernel_launch(void* const* d_in, const int* in_sizes, int n_in,
                              void* d_out, int out_size, void* d_ws, size_t ws_size,
                              hipStream_t stream) {
  (void)in_sizes; (void)n_in; (void)out_size; (void)ws_size;
  const float* x  = (const float*)d_in[0];
  const float* Wq = (const float*)d_in[1];
  const float* Wk = (const float*)d_in[2];
  const float* Wv = (const float*)d_in[3];
  const float* Wp = (const float*)d_in[4];
  const float* bp = (const float*)d_in[5];
  float* out = (float*)d_out;

  const size_t nx  = (size_t)BB * TT * CC;  // 6,291,456
  const size_t nw  = (size_t)HH * CC * HD;  //   147,456
  const size_t nwp = (size_t)CC * CC;       //   147,456

  _Float16* ws  = (_Float16*)d_ws;
  _Float16* xh  = ws;            ws += nx;
  _Float16* wqt = ws;            ws += nw;
  _Float16* wkt = ws;            ws += nw;
  _Float16* wvt = ws;            ws += nw;
  _Float16* wph = ws;            ws += nwp;
  _Float16* qh  = ws;            ws += nx;
  _Float16* kh  = ws;            ws += nx;
  _Float16* vt  = ws;            ws += nx;
  _Float16* ah  = ws;            ws += nx;   // ~61 MiB total

  k_cvt_f32_f16<<<(int)(nx / 256), 256, 0, stream>>>(x, xh, (int)nx);
  k_cvt_f32_f16<<<(int)(nwp / 256), 256, 0, stream>>>(Wp, wph, (int)nwp);
  k_transpose_w<<<(int)(nw / 256), 256, 0, stream>>>(Wq, wqt);
  k_transpose_w<<<(int)(nw / 256), 256, 0, stream>>>(Wk, wkt);
  k_transpose_w<<<(int)(nw / 256), 256, 0, stream>>>(Wv, wvt);

  k_qkv<<<BB * (TT / 16), 256, 0, stream>>>(xh, wqt, wkt, wvt, qh, kh, vt);
  k_attn<<<BB * HH * 2, 256, 0, stream>>>(qh, kh, vt, ah);
  k_proj<<<(BB * TT) / 16, 256, 0, stream>>>(ah, wph, bp, out);
}
